// Seq2Seq_Encoder_28509992911143
// MI455X (gfx1250) — matrix operand order repaired and checked by reading, not compiled, not run on hardware
//
#include <hip/hip_runtime.h>
#include <hip/hip_bf16.h>
#include <math.h>

// ---------------------------------------------------------------------------
// Seq2Seq encoder for MI455X (gfx1250, wave32, WMMA).
//   out[b, 0:256]   = h_last   (LN-GRU scan, 512 steps)
//   out[b, 256:512] = max_t conv1d(x_emb, kernel)   (K=3, VALID)
// ---------------------------------------------------------------------------

typedef __bf16 bf16_t;
typedef __attribute__((ext_vector_type(8)))  bf16_t v8bf;
typedef __attribute__((ext_vector_type(16))) bf16_t v16bf;
typedef __attribute__((ext_vector_type(8)))  float  v8f;

#define EPSV 1e-5f

// dims (fixed by the reference)
#define BB   64      // batch
#define TT   512     // time
#define UU   256     // units
#define N3   768     // 3*units
#define FF   256     // conv filters
#define S1T  (N3 * BB)   // 49152 floats per timestep in s1 (t-major, [t][col][b])

// ---------------- WMMA helpers (bf16 A/B, f32 acc, 16x16x32) ---------------

__device__ __forceinline__ v8f wmma_bf16(v16bf a, v16bf b, v8f c) {
  return __builtin_amdgcn_wmma_f32_16x16x32_bf16(false, a, false, b,
                                                 (short)0, c, false, false);
}

// A-matrix 16x32 bf16 fragment. Lane: m = lane&15, K-halves {h..h+7, 16+h..}
// with h = 8*(lane>=16). Two 16B loads.
__device__ __forceinline__ v16bf frag_a(const bf16_t* p, int stride, int m,
                                        int k0, int lane) {
  int half = (lane >> 4) * 8;
  const bf16_t* q = p + m * stride + k0 + half;
  v8bf lo = *(const v8bf*)(q);
  v8bf hi = *(const v8bf*)(q + 16);
  return __builtin_shufflevector(lo, hi, 0,1,2,3,4,5,6,7,8,9,10,11,12,13,14,15);
}

// B-matrix 32x16 bf16 fragment from transposed weights Wt[n][k].
// Lane: n = lane&15, K-halves {h..h+7, 16+h..} with h = 8*(lane>=16). Two 16B loads.
__device__ __forceinline__ v16bf frag_b(const bf16_t* p, int stride, int n,
                                        int k0, int lane) {
  int half = (lane >> 4) * 8;
  const bf16_t* q = p + n * stride + k0 + half;
  v8bf lo = *(const v8bf*)(q);
  v8bf hi = *(const v8bf*)(q + 16);
  return __builtin_shufflevector(lo, hi, 0,1,2,3,4,5,6,7,8,9,10,11,12,13,14,15);
}

// ------------------------- kernel 1: embedding gather ----------------------
// emb_bf16[(t*B + b)*256 + c], t padded to T+2 with zeros (conv halo).
__global__ void embed_kernel(const int* __restrict__ x,
                             const float* __restrict__ emb,
                             bf16_t* __restrict__ out) {
  int idx = blockIdx.x * 256 + threadIdx.x;
  const int total = (TT + 2) * BB * UU;
  if (idx >= total) return;
  int c  = idx % UU;
  int tb = idx / UU;
  int b  = tb % BB;
  int t  = tb / BB;
  float v = 0.0f;
  if (t < TT) {
    int tok = x[b * TT + t];
    v = emb[(long)tok * UU + c];
  }
  out[idx] = (bf16_t)v;
}

// -------------------- kernel 2: transpose + bf16 weights -------------------
// Wt[n][k] = W[k][n], Ut[n][k] = U[k][n] (n<768,k<256), kt[f][kk] = kernel[kk][f]
__global__ void convert_kernel(const float* __restrict__ W,
                               const float* __restrict__ Umat,
                               const float* __restrict__ kern,
                               bf16_t* __restrict__ Wt,
                               bf16_t* __restrict__ Ut,
                               bf16_t* __restrict__ kt) {
  int idx = blockIdx.x * 256 + threadIdx.x;
  const int seglen = N3 * UU;           // 196608
  if (idx >= 3 * seglen) return;
  int seg = idx / seglen, r = idx % seglen;
  if (seg == 0) {
    int n = r / UU, k = r % UU;
    Wt[r] = (bf16_t)W[k * N3 + n];
  } else if (seg == 1) {
    int n = r / UU, k = r % UU;
    Ut[r] = (bf16_t)Umat[k * N3 + n];
  } else {
    int f = r / N3, kk = r % N3;
    kt[r] = (bf16_t)kern[kk * FF + f];
  }
}

// --------------- kernel 3: s1 = LN(x_emb @ W + b; g0, beta0) ---------------
// One block = 16 rows (row = t*B + b, always inside one t), full 768 cols.
// Output layout is t-major [t][col][b] so the scan can read 8 consecutive
// batch rows per column with one 32B load.
__global__ void __launch_bounds__(256)
s1_gemm_ln(const bf16_t* __restrict__ embb, const bf16_t* __restrict__ Wt,
           const float* __restrict__ bvec, const float* __restrict__ g0,
           const float* __restrict__ b0, float* __restrict__ s1) {
  __shared__ bf16_t Al[16 * 272];
  __shared__ float  Yl[16 * 776];
  __shared__ float  ssum[16], ssq[16];

  int tid = threadIdx.x, lane = tid & 31, w = tid >> 5;
  int ln = lane & 15, rbase = (lane >> 4) * 8;
  long r0 = (long)blockIdx.x * 16;
  int  t  = (int)(r0 >> 6);          // 16 rows always within one timestep
  int  b0r = (int)(r0 & 63);         // first batch row of this block

  { // stage 16x256 A tile (32B per thread, 32B aligned)
    int row = tid >> 4, c0 = (tid & 15) * 16;
    *(v16bf*)&Al[row * 272 + c0] = *(const v16bf*)(embb + (r0 + row) * UU + c0);
  }
  __syncthreads();

  v16bf afr[8];
#pragma unroll
  for (int kc = 0; kc < 8; ++kc) afr[kc] = frag_a(Al, 272, ln, kc * 32, lane);

  for (int j = 0; j < 6; ++j) {
    int nt = w * 6 + j;
    v8f acc = {};
#pragma unroll
    for (int kc = 0; kc < 8; ++kc) {
      v16bf bf = frag_b(Wt, UU, nt * 16 + ln, kc * 32, lane);
      acc = wmma_bf16(afr[kc], bf, acc);
    }
    int col = nt * 16 + ln;
    float bias = bvec[col];
#pragma unroll
    for (int v = 0; v < 8; ++v) Yl[(rbase + v) * 776 + col] = acc[v] + bias;
  }
  __syncthreads();

  // per-row mean/var over 768 (wave w handles rows 2w, 2w+1)
  for (int rr = 0; rr < 2; ++rr) {
    int row = w * 2 + rr;
    float s = 0.f, q = 0.f;
    for (int i = lane; i < N3; i += 32) {
      float v = Yl[row * 776 + i];
      s += v; q += v * v;
    }
#pragma unroll
    for (int d = 16; d >= 1; d >>= 1) {
      s += __shfl_xor(s, d, 32);
      q += __shfl_xor(q, d, 32);
    }
    if (lane == 0) { ssum[row] = s; ssq[row] = q; }
  }
  __syncthreads();

  const float inv = 1.0f / (float)N3;
  float* s1t = s1 + (long)t * S1T;
  // enumerate (col, row) so 16 consecutive threads write 64B-contiguous runs
  for (int idx = tid; idx < 16 * N3; idx += 256) {
    int col = idx >> 4, row = idx & 15;
    float m   = ssum[row] * inv;
    float var = ssq[row] * inv - m * m;
    float st  = sqrtf(var + EPSV) + EPSV;
    float val = (Yl[row * 776 + col] - m) / st;
    s1t[col * BB + b0r + row] = g0[col] * val + b0[col];
  }
}

// ------------------------ kernel 4: persistent GRU scan --------------------
// 1 workgroup, 512 threads = 16 waves. h lives in registers; one LDS bf16
// buffer is time-multiplexed: h (phase1 A) -> r*h (phase2 A) -> h_next.
// Wave w owns columns [16w, 16w+16) of h; stats via shfl-butterfly + ds_add.
// s1 is [t][col][b]: one v8f load covers 8 batch rows of a tile.
__global__ void __launch_bounds__(512)
rnn_scan(const bf16_t* __restrict__ Ut, const float* __restrict__ s1,
         const int* __restrict__ x, const float* __restrict__ gam,
         const float* __restrict__ bet, float* __restrict__ out) {
  __shared__ bf16_t buf[64 * 272];          // 34 KB, h / rh / h_next
  __shared__ float s1sum[64], s1sq[64], s2sum[64], s2sq[64];
  __shared__ int   tokv[64];                // per-step mask tokens

  int tid = threadIdx.x, lane = tid & 31, w = tid >> 5;  // w in [0,16)
  int ln = lane & 15, rbase = (lane >> 4) * 8;
  int mycol = w * 16 + ln;                  // owned hidden column
  const float* g1 = gam + N3;               // gammas[1]
  const float* b1 = bet + N3;

  for (int i = tid; i < 64 * 272; i += 512) buf[i] = (bf16_t)0.0f;
  float hreg[4][8];
#pragma unroll
  for (int mt = 0; mt < 4; ++mt)
#pragma unroll
    for (int v = 0; v < 8; ++v) hreg[mt][v] = 0.0f;

  const float inv512 = 1.0f / 512.0f, inv256 = 1.0f / 256.0f;

  for (int t = 0; t < TT; ++t) {
    const float* s1t = s1 + (long)t * S1T;
    if (tid < 64) {
      s1sum[tid] = 0.f; s1sq[tid] = 0.f; s2sum[tid] = 0.f; s2sq[tid] = 0.f;
      tokv[tid] = x[tid * TT + t];
    }
    // prefetch next step's s1 slab (192KB) into cache while we compute
    if (t + 1 < TT) {
      const char* nxt = (const char*)(s1 + (long)(t + 1) * S1T) + tid * 384;
#pragma unroll
      for (int j = 0; j < 3; ++j) __builtin_prefetch(nxt + j * 128, 0, 3);
    }
    __syncthreads();                                   // h visible, stats zeroed

    // ---- phase 1: [z|r] = h @ U[:, :512]  (8 tiles/wave: nt=w and nt=16+w)
    v8f accz[4], accr[4];
#pragma unroll
    for (int mt = 0; mt < 4; ++mt) { accz[mt] = {}; accr[mt] = {}; }
#pragma unroll
    for (int kc = 0; kc < 8; ++kc) {
      v16bf bz = frag_b(Ut, UU, w * 16 + ln, kc * 32, lane);         // cols [0,256)
      v16bf br = frag_b(Ut, UU, 256 + w * 16 + ln, kc * 32, lane);   // cols [256,512)
#pragma unroll
      for (int mt = 0; mt < 4; ++mt) {
        v16bf a = frag_a(buf, 272, mt * 16 + ln, kc * 32, lane);
        accz[mt] = wmma_bf16(a, bz, accz[mt]);
        accr[mt] = wmma_bf16(a, br, accr[mt]);
      }
    }
    // LN stats over 512 cols (each lane contributes its z-col and r-col)
#pragma unroll
    for (int mt = 0; mt < 4; ++mt)
#pragma unroll
      for (int v = 0; v < 8; ++v) {
        float s = accz[mt][v] + accr[mt][v];
        float q = accz[mt][v] * accz[mt][v] + accr[mt][v] * accr[mt][v];
#pragma unroll
        for (int d = 1; d <= 8; d <<= 1) {             // 16-lane half reduce
          s += __shfl_xor(s, d, 32);
          q += __shfl_xor(q, d, 32);
        }
        if (ln == 0) {
          int row = mt * 16 + rbase + v;
          atomicAdd(&s1sum[row], s);                   // ds_add_f32
          atomicAdd(&s1sq[row], q);
        }
      }
    __syncthreads();                                   // buf reads done, stats ready

    // ---- z, r, and rh = r*h written over buf
    float zreg[4][8];
    float gz = g1[mycol],       bz2 = b1[mycol];
    float gr = g1[256 + mycol], br2 = b1[256 + mycol];
#pragma unroll
    for (int mt = 0; mt < 4; ++mt) {
      v8f s1z = *(const v8f*)(s1t + mycol * BB + mt * 16 + rbase);
      v8f s1r = *(const v8f*)(s1t + (256 + mycol) * BB + mt * 16 + rbase);
#pragma unroll
      for (int v = 0; v < 8; ++v) {
        int row = mt * 16 + rbase + v;
        float mean = s1sum[row] * inv512;
        float var  = s1sq[row] * inv512 - mean * mean;
        float st   = sqrtf(var + EPSV) + EPSV;
        float nz = (accz[mt][v] - mean) / st * gz + bz2;
        float sz = s1z[v] + nz;
        zreg[mt][v] = fminf(fmaxf(0.2f * sz + 0.5f, 0.f), 1.f);
        float nr = (accr[mt][v] - mean) / st * gr + br2;
        float sr = s1r[v] + nr;
        float r  = fminf(fmaxf(0.2f * sr + 0.5f, 0.f), 1.f);
        buf[row * 272 + mycol] = (bf16_t)(r * hreg[mt][v]);
      }
    }
    __syncthreads();                                   // rh visible

    // ---- phase 2: (r*h) @ U[:, 512:768]  (4 tiles/wave: nt2 = w)
    v8f acch[4];
#pragma unroll
    for (int mt = 0; mt < 4; ++mt) acch[mt] = {};
#pragma unroll
    for (int kc = 0; kc < 8; ++kc) {
      v16bf bh = frag_b(Ut, UU, 512 + w * 16 + ln, kc * 32, lane);
#pragma unroll
      for (int mt = 0; mt < 4; ++mt) {
        v16bf a = frag_a(buf, 272, mt * 16 + ln, kc * 32, lane);
        acch[mt] = wmma_bf16(a, bh, acch[mt]);
      }
    }
#pragma unroll
    for (int mt = 0; mt < 4; ++mt)
#pragma unroll
      for (int v = 0; v < 8; ++v) {
        float s = acch[mt][v], q = s * s;
#pragma unroll
        for (int d = 1; d <= 8; d <<= 1) {
          s += __shfl_xor(s, d, 32);
          q += __shfl_xor(q, d, 32);
        }
        if (ln == 0) {
          int row = mt * 16 + rbase + v;
          atomicAdd(&s2sum[row], s);
          atomicAdd(&s2sq[row], q);
        }
      }
    __syncthreads();                                   // rh reads done, stats2 ready

    // ---- hc = tanh(s1[:,512:] + LN2), blend, mask, write h_next over buf
    float gh = g1[512 + mycol], bh2 = b1[512 + mycol];
#pragma unroll
    for (int mt = 0; mt < 4; ++mt) {
      v8f s1h = *(const v8f*)(s1t + (512 + mycol) * BB + mt * 16 + rbase);
#pragma unroll
      for (int v = 0; v < 8; ++v) {
        int row = mt * 16 + rbase + v;
        float mean = s2sum[row] * inv256;
        float var  = s2sq[row] * inv256 - mean * mean;
        float st   = sqrtf(var + EPSV) + EPSV;
        float nh = (acch[mt][v] - mean) / st * gh + bh2;
        float hc = tanhf(s1h[v] + nh);
        float z  = zreg[mt][v];
        float hn = z * hreg[mt][v] + (1.0f - z) * hc;
        float ho = (tokv[row] != 0) ? hn : hreg[mt][v];
        hreg[mt][v] = ho;
        buf[row * 272 + mycol] = (bf16_t)ho;
      }
    }
    __syncthreads();
  }

#pragma unroll
  for (int mt = 0; mt < 4; ++mt)
#pragma unroll
    for (int v = 0; v < 8; ++v)
      out[(mt * 16 + rbase + v) * 512 + mycol] = hreg[mt][v];
}

// ------------------ kernel 5: conv1d (K=3, VALID) + max over t -------------
// block = one batch b; 8 waves x 2 f-tiles. K = 3*256 = 768 via im2col-free
// A fragments read straight from padded emb_bf16[t][b][c].
__global__ void __launch_bounds__(256)
conv_max(const bf16_t* __restrict__ embb, const bf16_t* __restrict__ kt,
         float* __restrict__ out) {
  int b = blockIdx.x;
  int tid = threadIdx.x, lane = tid & 31, w = tid >> 5;
  int ln = lane & 15, rbase = (lane >> 4) * 8;
  float mx[2][8];
#pragma unroll
  for (int j = 0; j < 2; ++j)
#pragma unroll
    for (int v = 0; v < 8; ++v) mx[j][v] = -3.0e38f;

  for (int t0 = 0; t0 < TT; t0 += 16) {
    v8f acc0 = {}, acc1 = {};
#pragma unroll
    for (int kc = 0; kc < 24; ++kc) {
      int k = kc >> 3, c0 = (kc & 7) * 32;
      const bf16_t* ap = embb + (((long)(t0 + ln + k) * BB + b) * UU + c0 + rbase);
      v8bf lo = *(const v8bf*)(ap);
      v8bf hi = *(const v8bf*)(ap + 16);
      v16bf a = __builtin_shufflevector(lo, hi,
                  0,1,2,3,4,5,6,7,8,9,10,11,12,13,14,15);
      v16bf f0 = frag_b(kt, N3, (2 * w) * 16 + ln, kc * 32, lane);
      v16bf f1 = frag_b(kt, N3, (2 * w + 1) * 16 + ln, kc * 32, lane);
      acc0 = wmma_bf16(a, f0, acc0);
      acc1 = wmma_bf16(a, f1, acc1);
    }
#pragma unroll
    for (int v = 0; v < 8; ++v) {
      int tt = t0 + rbase + v;
      bool ok = tt < (TT - 2);                         // 510 valid positions
      mx[0][v] = fmaxf(mx[0][v], ok ? acc0[v] : -3.0e38f);
      mx[1][v] = fmaxf(mx[1][v], ok ? acc1[v] : -3.0e38f);
    }
  }
#pragma unroll
  for (int j = 0; j < 2; ++j) {
    float m = mx[j][0];
#pragma unroll
    for (int v = 1; v < 8; ++v) m = fmaxf(m, mx[j][v]);
    m = fmaxf(m, __shfl_xor(m, 16, 32));               // merge row halves
    if (lane < 16) out[b * 512 + 256 + (2 * w + j) * 16 + ln] = m;
  }
}

// ---------------------------------- launcher -------------------------------
extern "C" void kernel_launch(void* const* d_in, const int* in_sizes, int n_in,
                              void* d_out, int out_size, void* d_ws, size_t ws_size,
                              hipStream_t stream) {
  const int*   x      = (const int*)d_in[0];
  // d_in[1] = bucket_size (== 512, baked in)
  const float* emb    = (const float*)d_in[2];
  const float* W      = (const float*)d_in[3];
  const float* Umat   = (const float*)d_in[4];
  const float* bvec   = (const float*)d_in[5];
  const float* gammas = (const float*)d_in[6];
  const float* betas  = (const float*)d_in[7];
  const float* kern   = (const float*)d_in[8];
  float* out = (float*)d_out;

  char* ws = (char*)d_ws;
  size_t off = 0;
  auto take = [&](size_t bytes) { char* p = ws + off; off = (off + bytes + 255) & ~(size_t)255; return p; };
  bf16_t* embb = (bf16_t*)take((size_t)(TT + 2) * BB * UU * 2);  // 16.8 MB
  bf16_t* Wt   = (bf16_t*)take((size_t)N3 * UU * 2);             // 384 KB
  bf16_t* Ut   = (bf16_t*)take((size_t)N3 * UU * 2);
  bf16_t* kt   = (bf16_t*)take((size_t)FF * N3 * 2);
  float*  s1   = (float*)take((size_t)TT * S1T * 4);             // 100.7 MB

  int embTotal = (TT + 2) * BB * UU;
  embed_kernel<<<(embTotal + 255) / 256, 256, 0, stream>>>(x, emb, embb);
  int cvtTotal = 3 * N3 * UU;
  convert_kernel<<<(cvtTotal + 255) / 256, 256, 0, stream>>>(W, Umat, kern, Wt, Ut, kt);
  s1_gemm_ln<<<(TT * BB) / 16, 256, 0, stream>>>(embb, Wt, bvec, gammas, betas, s1);
  rnn_scan<<<1, 512, 0, stream>>>(Ut, s1, x, gammas, betas, out);
  conv_max<<<BB, 256, 0, stream>>>(embb, kt, out);
}
